// CTM_30846455120449
// MI455X (gfx1250) — matrix operand order repaired and checked by reading, not compiled, not run on hardware
//
#include <hip/hip_runtime.h>
#include <hip/hip_bf16.h>
#include <math.h>

#define KDIM 512
#define RHO_C 0.01f

typedef __attribute__((ext_vector_type(16))) _Float16 v16h;
typedef __attribute__((ext_vector_type(8)))  float    v8f;

// ---------------- theta = softmax(alpha) ----------------
__global__ void ctm_theta(const float* __restrict__ alpha, float* __restrict__ theta) {
    __shared__ float red[512];
    int tid = threadIdx.x;
    float x = alpha[tid];
    red[tid] = x; __syncthreads();
    for (int s = 256; s > 0; s >>= 1) { if (tid < s) red[tid] = fmaxf(red[tid], red[tid+s]); __syncthreads(); }
    float mx = red[0]; __syncthreads();
    float e = __expf(x - mx);
    red[tid] = e; __syncthreads();
    for (int s = 256; s > 0; s >>= 1) { if (tid < s) red[tid] += red[tid+s]; __syncthreads(); }
    theta[tid] = e / red[0];
}

// ---------------- B = softmax(beta, axis=1) ----------------
__global__ void ctm_beta_softmax(const float* __restrict__ beta, float* __restrict__ B) {
    __shared__ float red[256];
    int row = blockIdx.x, tid = threadIdx.x;
    float x0 = beta[row*KDIM + tid], x1 = beta[row*KDIM + tid + 256];
    red[tid] = fmaxf(x0, x1); __syncthreads();
    for (int s = 128; s > 0; s >>= 1) { if (tid < s) red[tid] = fmaxf(red[tid], red[tid+s]); __syncthreads(); }
    float mx = red[0]; __syncthreads();
    float e0 = __expf(x0 - mx), e1 = __expf(x1 - mx);
    red[tid] = e0 + e1; __syncthreads();
    for (int s = 128; s > 0; s >>= 1) { if (tid < s) red[tid] += red[tid+s]; __syncthreads(); }
    float inv = 1.f / red[0];
    B[row*KDIM + tid] = e0 * inv;
    B[row*KDIM + tid + 256] = e1 * inv;
}

// ---------------- L = cholesky(sigma), general, single workgroup ----------------
__global__ __launch_bounds__(1024) void ctm_chol(const float* __restrict__ sigma, float* __restrict__ L) {
    int tid = threadIdx.x;
    for (int idx = tid; idx < KDIM*KDIM; idx += 1024) {
        int i = idx >> 9, j = idx & 511;
        L[idx] = (j <= i) ? sigma[idx] : 0.f;
    }
    __syncthreads();
    for (int k = 0; k < KDIM; ++k) {
        float s = L[k*KDIM + k];
        float d = sqrtf(fmaxf(s, 1e-30f));
        float invd = 1.f / d;
        __syncthreads();                      // all read diag before overwrite
        if (tid == 0) L[k*KDIM + k] = d;
        for (int i = k+1+tid; i < KDIM; i += 1024) L[i*KDIM + k] *= invd;
        __syncthreads();
        int cnt = KDIM - 1 - k;
        int total = cnt * cnt;
        for (int idx = tid; idx < total; idx += 1024) {
            int i = k + 1 + idx / cnt;
            int j = k + 1 + idx % cnt;
            if (j <= i) L[i*KDIM + j] -= L[i*KDIM + k] * L[j*KDIM + k];
        }
        __syncthreads();
    }
}

// ---------------- r = mu @ B ----------------
__global__ void ctm_rvec(const float* __restrict__ mu, const float* __restrict__ B, float* __restrict__ rv) {
    int n = blockIdx.x * blockDim.x + threadIdx.x;
    if (n < KDIM) {
        float s = 0.f;
        for (int t = 0; t < KDIM; ++t) s += mu[t] * B[t*KDIM + n];
        rv[n] = s;
    }
}

// ---------------- M = L^T @ B, stored f16 in WMMA B-operand lane layout ----------------
// B-operand (16-bit, 32x16 tile): lanes 0-15 hold K{0..7,16..23}, col = lane;
// lanes 16-31 hold K{8..15,24..31}, col = lane-16. Tiles ordered nt-major so the kt loop is contiguous.
__global__ void ctm_pack(const float* __restrict__ L, const float* __restrict__ B, _Float16* __restrict__ Mh) {
    int idx = blockIdx.x * blockDim.x + threadIdx.x;   // 0 .. 512*512-1
    int k = idx >> 9, n = idx & 511;
    float s = 0.f;
    for (int t = k; t < KDIM; ++t)                      // L lower-triangular: col k nonzero for t>=k
        s += L[t*KDIM + k] * B[t*KDIM + n];
    int kt = k >> 5, nt = n >> 4, kk = k & 31;
    int lane = ((kk & 8) << 1) + (n & 15);
    int h    = (kk & 7) | ((kk & 16) >> 1);
    size_t off = (((size_t)(nt*16 + kt) * 32) + lane) * 16 + h;
    Mh[off] = (_Float16)s;
}

__device__ __forceinline__ v16h cvt_a_tile(const float* ap) {
    float4 f0 = *(const float4*)(ap);
    float4 f1 = *(const float4*)(ap + 4);
    float4 f2 = *(const float4*)(ap + 16);
    float4 f3 = *(const float4*)(ap + 20);
    v16h a;
    a[0]=(_Float16)f0.x;  a[1]=(_Float16)f0.y;  a[2]=(_Float16)f0.z;  a[3]=(_Float16)f0.w;
    a[4]=(_Float16)f1.x;  a[5]=(_Float16)f1.y;  a[6]=(_Float16)f1.z;  a[7]=(_Float16)f1.w;
    a[8]=(_Float16)f2.x;  a[9]=(_Float16)f2.y;  a[10]=(_Float16)f2.z; a[11]=(_Float16)f2.w;
    a[12]=(_Float16)f3.x; a[13]=(_Float16)f3.y; a[14]=(_Float16)f3.z; a[15]=(_Float16)f3.w;
    return a;
}

// ---------------- main: gamma = normalize(softmax(r + eps@M)*theta + rho) ----------------
// 4 waves / block, 32 rows (two 16-row A tiles) x 512 cols; each B tile feeds 2 WMMAs.
__global__ __launch_bounds__(128) void ctm_main(
    const float* __restrict__ eps, const _Float16* __restrict__ Mh,
    const float* __restrict__ theta, const float* __restrict__ rv,
    float* __restrict__ out)
{
    __shared__ float Lg[16][516];      // padded rows: 516%64==4 banks offset
    __shared__ float thetaL[KDIM];
    __shared__ float rvL[KDIM];
    __shared__ float red[128];
    __shared__ float invA[16];
    __shared__ float totA[16];

    const int tid  = threadIdx.x;
    const int lane = tid & 31;
    const int w    = tid >> 5;         // wave 0..3, owns 8 column tiles
    const int r0   = blockIdx.x * 32;

    for (int i = tid; i < KDIM; i += 128) { thetaL[i] = theta[i]; rvL[i] = rv[i]; }

    v8f acc[2][8];
    v8f zero = {0.f,0.f,0.f,0.f,0.f,0.f,0.f,0.f};
#pragma unroll
    for (int p = 0; p < 2; ++p)
#pragma unroll
        for (int t = 0; t < 8; ++t) acc[p][t] = zero;

    // A-operand layout (16-bit 16x32): lane<16 -> row=lane, halves = K{0..7,16..23};
    // lane>=16 -> row=lane-16, halves = K{8..15,24..31}.
    const int arow = r0 + (lane & 15);
    const int koff = (lane >> 4) << 3;                 // 0 or 8

    for (int kt = 0; kt < 16; ++kt) {
        const float* ap0 = eps + (size_t)arow * KDIM + kt*32 + koff;
        v16h a0 = cvt_a_tile(ap0);
        v16h a1 = cvt_a_tile(ap0 + 16*KDIM);
#pragma unroll
        for (int t = 0; t < 8; ++t) {
            int tileIdx = (w*8 + t)*16 + kt;
            v16h b = *(const v16h*)(Mh + (size_t)tileIdx * 512 + lane * 16);
            acc[0][t] = __builtin_amdgcn_wmma_f32_16x16x32_f16(
                false, a0, false, b, (short)0, acc[0][t], false, false);
            acc[1][t] = __builtin_amdgcn_wmma_f32_16x16x32_f16(
                false, a1, false, b, (short)0, acc[1][t], false, false);
        }
    }

    // C/D layout: VGPR v, lanes 0-15: M=v, N=lane; lanes 16-31: M=v+8, N=lane-16
    const int mbase  = (lane >> 4) << 3;
    const int nlocal = lane & 15;
    const int row = tid >> 3;          // epilogue mapping: 8 threads per row
    const int s8  = tid & 7;
    const int c0  = s8 * 64;

    for (int p = 0; p < 2; ++p) {
        __syncthreads();               // preload visible (p=0); prior pass done with Lg (p=1)
#pragma unroll
        for (int t = 0; t < 8; ++t) {
            int col = (w*8 + t)*16 + nlocal;
            float rvc = rvL[col];
#pragma unroll
            for (int v = 0; v < 8; ++v)
                Lg[v + mbase][col] = acc[p][t][v] + rvc;
        }
        __syncthreads();

        // row-wise softmax + gamma normalization over 16 rows
        float m = -1e30f;
        for (int c = c0; c < c0 + 64; ++c) m = fmaxf(m, Lg[row][c]);
        red[tid] = m; __syncthreads();
        float rowmax = -1e30f;
#pragma unroll
        for (int i = 0; i < 8; ++i) rowmax = fmaxf(rowmax, red[row*8 + i]);
        __syncthreads();

        float s = 0.f;
        for (int c = c0; c < c0 + 64; ++c) {
            float e = __expf(Lg[row][c] - rowmax);
            Lg[row][c] = e;
            s += e;
        }
        red[tid] = s; __syncthreads();
        float rowsum = 0.f;
#pragma unroll
        for (int i = 0; i < 8; ++i) rowsum += red[row*8 + i];
        float inv = 1.f / rowsum;
        __syncthreads();

        float g = 0.f;
        for (int c = c0; c < c0 + 64; ++c) g += Lg[row][c] * inv * thetaL[c] + RHO_C;
        red[tid] = g; __syncthreads();
        float tot = 0.f;
#pragma unroll
        for (int i = 0; i < 8; ++i) tot += red[row*8 + i];
        if (s8 == 0) { invA[row] = inv; totA[row] = 1.f / tot; }
        __syncthreads();

        // coalesced writes: consecutive tid -> consecutive column
        const size_t rbase = (size_t)(r0 + p*16) * KDIM;
#pragma unroll 4
        for (int mrow = 0; mrow < 16; ++mrow) {
            float ia = invA[mrow], ta = totA[mrow];
#pragma unroll
            for (int j = 0; j < 4; ++j) {
                int c = tid + j*128;
                out[rbase + (size_t)mrow * KDIM + c] =
                    (Lg[mrow][c] * ia * thetaL[c] + RHO_C) * ta;
            }
        }
    }
}

extern "C" void kernel_launch(void* const* d_in, const int* in_sizes, int n_in,
                              void* d_out, int out_size, void* d_ws, size_t ws_size,
                              hipStream_t stream) {
    (void)in_sizes; (void)n_in; (void)out_size; (void)ws_size;
    // inputs: 0=bow(int, unused), 1=alpha[512], 2=beta[512*512], 3=sigma[512*512], 4=mu[512], 5=eps[N*512]
    const float* alpha = (const float*)d_in[1];
    const float* beta  = (const float*)d_in[2];
    const float* sigma = (const float*)d_in[3];
    const float* mu    = (const float*)d_in[4];
    const float* eps   = (const float*)d_in[5];
    float* out = (float*)d_out;

    char* ws = (char*)d_ws;
    float*    Bm    = (float*)(ws);                                   // 1 MB
    float*    Lm    = (float*)(ws + (1u<<20));                        // 1 MB
    _Float16* Mh    = (_Float16*)(ws + (2u<<20));                     // 512 KB
    float*    theta = (float*)(ws + (2u<<20) + KDIM*KDIM*2);          // 2 KB
    float*    rv    = theta + KDIM;                                   // 2 KB

    ctm_theta<<<1, 512, 0, stream>>>(alpha, theta);
    ctm_beta_softmax<<<KDIM, 256, 0, stream>>>(beta, Bm);
    ctm_chol<<<1, 1024, 0, stream>>>(sigma, Lm);
    ctm_rvec<<<2, 256, 0, stream>>>(mu, Bm, rv);
    ctm_pack<<<(KDIM*KDIM)/256, 256, 0, stream>>>(Lm, Bm, Mh);

    const int N = 131072;
    ctm_main<<<N/32, 128, 0, stream>>>(eps, Mh, theta, rv, out);
}
